// PatchReservoir_74981539053643
// MI455X (gfx1250) — matrix operand order repaired and checked by reading, not compiled, not run on hardware
//
#include <hip/hip_runtime.h>
#include <math.h>

// ---------------- problem constants ----------------
#define IN_CH   3
#define PATCH   64
#define CSTRIDE 32
#define RES     4096
#define NCLS    1000
#define BATCH   128
#define LSTEPS  256
#define XLEN    8224
#define KCAT    (RES + PATCH)   // 4160: [Wres ; WinT] stacked along K
#define NPAD    1024            // classifier N padded to WMMA tiles

typedef __attribute__((ext_vector_type(16))) __bf16 v16bf;
typedef __attribute__((ext_vector_type(8)))  __bf16 v8bf;
typedef __attribute__((ext_vector_type(8)))  float  v8f;

static __device__ __forceinline__ __bf16 f2bf(float f) { return (__bf16)f; }

// Load the two contiguous 8x-bf16 (128-bit) halves of a WMMA operand lane.
static __device__ __forceinline__ v16bf load_pair(const __bf16* p0, const __bf16* p1) {
  v8bf lo = *(const v8bf*)p0;
  v8bf hi = *(const v8bf*)p1;
  v16bf r;
#pragma unroll
  for (int i = 0; i < 8; ++i) { r[i] = lo[i]; r[i + 8] = hi[i]; }
  return r;
}

// ---------------- prep kernels ----------------

// Conv1d patch embed -> featU[t][b][oc] (bf16), t in [0,256)
__global__ __launch_bounds__(256)
void conv_kernel(const float* __restrict__ x, const float* __restrict__ w,
                 const float* __restrict__ bias, __bf16* __restrict__ featU) {
  int idx = blockIdx.x * 256 + threadIdx.x;          // ((t*BATCH)+b)*PATCH+oc
  if (idx >= LSTEPS * BATCH * PATCH) return;
  int oc = idx & (PATCH - 1);
  int b  = (idx >> 6) & (BATCH - 1);
  int t  = idx >> 13;
  float s = bias[oc];
  const float* xb = x + (size_t)b * IN_CH * XLEN + t * CSTRIDE;
  const float* wo = w + (size_t)oc * IN_CH * PATCH;
#pragma unroll
  for (int ic = 0; ic < IN_CH; ++ic) {
    const float* xi = xb + (size_t)ic * XLEN;
    const float* wi = wo + ic * PATCH;
#pragma unroll 8
    for (int kk = 0; kk < PATCH; ++kk) s += xi[kk] * wi[kk];
  }
  featU[idx] = f2bf(s);
}

// Pack Wt[n][k] = (k<RES ? Wres[k][n] : W_input[n][k-RES]) as bf16, K contiguous.
__global__ __launch_bounds__(256)
void pack_w_kernel(const float* __restrict__ Wres, const float* __restrict__ Win,
                   __bf16* __restrict__ Wt) {
  size_t idx = (size_t)blockIdx.x * 256 + threadIdx.x;
  if (idx >= (size_t)RES * KCAT) return;
  int k = (int)(idx % KCAT);
  int n = (int)(idx / KCAT);
  float v = (k < RES) ? Wres[(size_t)k * RES + n]
                      : Win[(size_t)n * PATCH + (k - RES)];
  Wt[idx] = f2bf(v);
}

// w_cls/w_dist are already [N][K] row-major; convert to bf16, pad rows to 1024.
__global__ __launch_bounds__(256)
void pack_cls_kernel(const float* __restrict__ wc, const float* __restrict__ wd,
                     __bf16* __restrict__ pc, __bf16* __restrict__ pd) {
  size_t idx = (size_t)blockIdx.x * 256 + threadIdx.x;
  if (idx >= (size_t)NPAD * RES) return;
  bool valid = idx < (size_t)NCLS * RES;
  pc[idx] = f2bf(valid ? wc[idx] : 0.0f);
  pd[idx] = f2bf(valid ? wd[idx] : 0.0f);
}

// Broadcast cls/dist tokens to [BATCH][PATCH] bf16
__global__ __launch_bounds__(256)
void fill_tok_kernel(const float* __restrict__ ct, const float* __restrict__ dt,
                     __bf16* __restrict__ uc, __bf16* __restrict__ ud) {
  int idx = blockIdx.x * 256 + threadIdx.x;
  if (idx >= BATCH * PATCH) return;
  int j = idx & (PATCH - 1);
  uc[idx] = f2bf(ct[j]);
  ud[idx] = f2bf(dt[j]);
}

__global__ __launch_bounds__(256)
void zero_h_kernel(__bf16* __restrict__ h) {
  int idx = blockIdx.x * 256 + threadIdx.x;
  if (idx < BATCH * RES) h[idx] = f2bf(0.0f);
}

// ---------------- reservoir step: h_out = tanh(h_in @ Wres + u @ WinT) ----------------
// Block = 8 waves, each wave: one 16-row M tile x 64-col N slab (4 WMMA accumulators).
// Grid.x = 64 (N slabs of 64 cover 4096). K = 4096 (h) + 64 (u).
__global__ __launch_bounds__(256)
void step_kernel(const __bf16* __restrict__ hin, const __bf16* __restrict__ u,
                 const __bf16* __restrict__ Wt, __bf16* __restrict__ hout) {
  const int lane = threadIdx.x & 31;
  const int wave = threadIdx.x >> 5;
  const int half = lane >> 4;        // K-half selector per WMMA lane layout
  const int lL   = lane & 15;
  const int m0   = wave * 16;
  const int n0   = blockIdx.x * 64;

  v8f acc0 = {}, acc1 = {}, acc2 = {}, acc3 = {};

  const __bf16* Arow = hin + (size_t)(m0 + lL) * RES;
  const __bf16* B0 = Wt + (size_t)(n0 +  0 + lL) * KCAT + half * 16;
  const __bf16* B1 = Wt + (size_t)(n0 + 16 + lL) * KCAT + half * 16;
  const __bf16* B2 = Wt + (size_t)(n0 + 32 + lL) * KCAT + half * 16;
  const __bf16* B3 = Wt + (size_t)(n0 + 48 + lL) * KCAT + half * 16;

  for (int kt = 0; kt < RES / 32; ++kt) {
    const int k0 = kt * 32;
    v16bf a  = load_pair(Arow + k0 + half * 16, Arow + k0 + half * 16 + 8);
    v16bf b0 = load_pair(B0 + k0, B0 + k0 + 8);
    v16bf b1 = load_pair(B1 + k0, B1 + k0 + 8);
    v16bf b2 = load_pair(B2 + k0, B2 + k0 + 8);
    v16bf b3 = load_pair(B3 + k0, B3 + k0 + 8);
    acc0 = __builtin_amdgcn_wmma_f32_16x16x32_bf16(false, a, false, b0, (short)0, acc0, false, false);
    acc1 = __builtin_amdgcn_wmma_f32_16x16x32_bf16(false, a, false, b1, (short)0, acc1, false, false);
    acc2 = __builtin_amdgcn_wmma_f32_16x16x32_bf16(false, a, false, b2, (short)0, acc2, false, false);
    acc3 = __builtin_amdgcn_wmma_f32_16x16x32_bf16(false, a, false, b3, (short)0, acc3, false, false);
  }

  // input-projection part: K rows [RES, RES+64) of Wt, A from u (row stride PATCH)
  const __bf16* Urow = u + (size_t)(m0 + lL) * PATCH;
#pragma unroll
  for (int kt = 0; kt < PATCH / 32; ++kt) {
    const int k0 = kt * 32;
    const int kw = RES + k0;
    v16bf a  = load_pair(Urow + k0 + half * 16, Urow + k0 + half * 16 + 8);
    v16bf b0 = load_pair(B0 + kw, B0 + kw + 8);
    v16bf b1 = load_pair(B1 + kw, B1 + kw + 8);
    v16bf b2 = load_pair(B2 + kw, B2 + kw + 8);
    v16bf b3 = load_pair(B3 + kw, B3 + kw + 8);
    acc0 = __builtin_amdgcn_wmma_f32_16x16x32_bf16(false, a, false, b0, (short)0, acc0, false, false);
    acc1 = __builtin_amdgcn_wmma_f32_16x16x32_bf16(false, a, false, b1, (short)0, acc1, false, false);
    acc2 = __builtin_amdgcn_wmma_f32_16x16x32_bf16(false, a, false, b2, (short)0, acc2, false, false);
    acc3 = __builtin_amdgcn_wmma_f32_16x16x32_bf16(false, a, false, b3, (short)0, acc3, false, false);
  }

#pragma unroll
  for (int r = 0; r < 8; ++r) {
    const int row = m0 + r + half * 8;
    __bf16* dst = hout + (size_t)row * RES + n0 + lL;
    dst[ 0] = f2bf(tanhf(acc0[r]));
    dst[16] = f2bf(tanhf(acc1[r]));
    dst[32] = f2bf(tanhf(acc2[r]));
    dst[48] = f2bf(tanhf(acc3[r]));
  }
}

// ---------------- classifier: 0.5*(h_cls@w_cls^T + b_cls + h_dist@w_dist^T + b_dist) ----------------
__global__ __launch_bounds__(256)
void cls_kernel(const __bf16* __restrict__ hc, const __bf16* __restrict__ hd,
                const __bf16* __restrict__ Wc, const __bf16* __restrict__ Wd,
                const float* __restrict__ bc, const float* __restrict__ bd,
                float* __restrict__ logits) {
  const int lane = threadIdx.x & 31;
  const int wave = threadIdx.x >> 5;
  const int half = lane >> 4;
  const int lL   = lane & 15;
  const int m0   = wave * 16;
  const int n0   = blockIdx.x * 64;

  v8f ac[4] = {}, ad[4] = {};
  const __bf16* Ac = hc + (size_t)(m0 + lL) * RES;
  const __bf16* Ad = hd + (size_t)(m0 + lL) * RES;
  const __bf16* Bc[4];
  const __bf16* Bd[4];
#pragma unroll
  for (int j = 0; j < 4; ++j) {
    Bc[j] = Wc + (size_t)(n0 + 16 * j + lL) * RES + half * 16;
    Bd[j] = Wd + (size_t)(n0 + 16 * j + lL) * RES + half * 16;
  }

  for (int kt = 0; kt < RES / 32; ++kt) {
    const int k0 = kt * 32;
    v16bf a1 = load_pair(Ac + k0 + half * 16, Ac + k0 + half * 16 + 8);
    v16bf a2 = load_pair(Ad + k0 + half * 16, Ad + k0 + half * 16 + 8);
#pragma unroll
    for (int j = 0; j < 4; ++j) {
      v16bf b1 = load_pair(Bc[j] + k0, Bc[j] + k0 + 8);
      v16bf b2 = load_pair(Bd[j] + k0, Bd[j] + k0 + 8);
      ac[j] = __builtin_amdgcn_wmma_f32_16x16x32_bf16(false, a1, false, b1, (short)0, ac[j], false, false);
      ad[j] = __builtin_amdgcn_wmma_f32_16x16x32_bf16(false, a2, false, b2, (short)0, ad[j], false, false);
    }
  }

#pragma unroll
  for (int j = 0; j < 4; ++j) {
    const int col = n0 + 16 * j + lL;
    if (col < NCLS) {
      const float bias = bc[col] + bd[col];
#pragma unroll
      for (int r = 0; r < 8; ++r) {
        const int row = m0 + r + half * 8;
        logits[(size_t)row * NPAD + col] = 0.5f * (ac[j][r] + ad[j][r] + bias);
      }
    }
  }
}

// ---------------- log_softmax over 1000 classes, one block per batch row ----------------
__global__ __launch_bounds__(256)
void logsoftmax_kernel(const float* __restrict__ logits, float* __restrict__ out) {
  __shared__ float red[256];
  const int b = blockIdx.x;
  const float* row = logits + (size_t)b * NPAD;
  float mx = -INFINITY;
  for (int c = threadIdx.x; c < NCLS; c += 256) mx = fmaxf(mx, row[c]);
  red[threadIdx.x] = mx; __syncthreads();
  for (int s = 128; s > 0; s >>= 1) {
    if (threadIdx.x < s) red[threadIdx.x] = fmaxf(red[threadIdx.x], red[threadIdx.x + s]);
    __syncthreads();
  }
  mx = red[0]; __syncthreads();
  float sum = 0.0f;
  for (int c = threadIdx.x; c < NCLS; c += 256) sum += expf(row[c] - mx);
  red[threadIdx.x] = sum; __syncthreads();
  for (int s = 128; s > 0; s >>= 1) {
    if (threadIdx.x < s) red[threadIdx.x] += red[threadIdx.x + s];
    __syncthreads();
  }
  const float lse = logf(red[0]) + mx;
  for (int c = threadIdx.x; c < NCLS; c += 256)
    out[(size_t)b * NCLS + c] = row[c] - lse;
}

// ---------------- workspace layout (bytes, all 256B aligned) ----------------
static constexpr size_t WT_OFF    = 0;                                   // 4096*4160*2
static constexpr size_t FEAT_OFF  = WT_OFF    + (size_t)RES * KCAT * 2;  // 256*128*64*2
static constexpr size_t HA_OFF    = FEAT_OFF  + (size_t)LSTEPS * BATCH * PATCH * 2;
static constexpr size_t HB_OFF    = HA_OFF    + (size_t)BATCH * RES * 2;
static constexpr size_t HCLS_OFF  = HB_OFF    + (size_t)BATCH * RES * 2;
static constexpr size_t HDIST_OFF = HCLS_OFF  + (size_t)BATCH * RES * 2;
static constexpr size_t UCLS_OFF  = HDIST_OFF + (size_t)BATCH * RES * 2;
static constexpr size_t UDIST_OFF = UCLS_OFF  + (size_t)BATCH * PATCH * 2;
static constexpr size_t WC_OFF    = UDIST_OFF + (size_t)BATCH * PATCH * 2;
static constexpr size_t WD_OFF    = WC_OFF    + (size_t)NPAD * RES * 2;
static constexpr size_t LOG_OFF   = WD_OFF    + (size_t)NPAD * RES * 2;  // 128*1024*4 f32

extern "C" void kernel_launch(void* const* d_in, const int* in_sizes, int n_in,
                              void* d_out, int out_size, void* d_ws, size_t ws_size,
                              hipStream_t stream) {
  const float* x      = (const float*)d_in[0];
  const float* conv_w = (const float*)d_in[1];
  const float* conv_b = (const float*)d_in[2];
  const float* Wres   = (const float*)d_in[3];
  const float* Win    = (const float*)d_in[4];
  const float* clsT   = (const float*)d_in[5];
  const float* distT  = (const float*)d_in[6];
  const float* w_cls  = (const float*)d_in[7];
  const float* b_cls  = (const float*)d_in[8];
  const float* w_dist = (const float*)d_in[9];
  const float* b_dist = (const float*)d_in[10];
  float* out = (float*)d_out;

  char* ws = (char*)d_ws;
  __bf16* Wt    = (__bf16*)(ws + WT_OFF);
  __bf16* featU = (__bf16*)(ws + FEAT_OFF);
  __bf16* bufs[2] = { (__bf16*)(ws + HA_OFF), (__bf16*)(ws + HB_OFF) };
  __bf16* hCls  = (__bf16*)(ws + HCLS_OFF);
  __bf16* hDist = (__bf16*)(ws + HDIST_OFF);
  __bf16* uCls  = (__bf16*)(ws + UCLS_OFF);
  __bf16* uDist = (__bf16*)(ws + UDIST_OFF);
  __bf16* Wc    = (__bf16*)(ws + WC_OFF);
  __bf16* Wd    = (__bf16*)(ws + WD_OFF);
  float*  logits = (float*)(ws + LOG_OFF);

  // ---- prep ----
  conv_kernel<<<(LSTEPS * BATCH * PATCH) / 256, 256, 0, stream>>>(x, conv_w, conv_b, featU);
  pack_w_kernel<<<(unsigned)(((size_t)RES * KCAT + 255) / 256), 256, 0, stream>>>(Wres, Win, Wt);
  pack_cls_kernel<<<(unsigned)(((size_t)NPAD * RES) / 256), 256, 0, stream>>>(w_cls, w_dist, Wc, Wd);
  fill_tok_kernel<<<(BATCH * PATCH) / 256, 256, 0, stream>>>(clsT, distT, uCls, uDist);
  zero_h_kernel<<<(BATCH * RES) / 256, 256, 0, stream>>>(bufs[0]);

  // ---- sequential reservoir scan: 256 WMMA GEMM+tanh steps ----
  for (int t = 0; t < LSTEPS; ++t) {
    step_kernel<<<64, 256, 0, stream>>>(bufs[t & 1],
                                        featU + (size_t)t * BATCH * PATCH,
                                        Wt, bufs[(t + 1) & 1]);
  }
  __bf16* hfin = bufs[LSTEPS & 1];  // LSTEPS even -> bufs[0]

  // ---- cls / dist token applications ----
  step_kernel<<<64, 256, 0, stream>>>(hfin, uCls,  Wt, hCls);
  step_kernel<<<64, 256, 0, stream>>>(hfin, uDist, Wt, hDist);

  // ---- classifier + log_softmax ----
  cls_kernel<<<NPAD / 64, 256, 0, stream>>>(hCls, hDist, Wc, Wd, b_cls, b_dist, logits);
  logsoftmax_kernel<<<BATCH, 256, 0, stream>>>(logits, out);
}
